// GAT_1709396983866
// MI455X (gfx1250) — matrix operand order repaired and checked by reading, not compiled, not run on hardware
//
#include <hip/hip_runtime.h>

typedef __attribute__((ext_vector_type(16))) __bf16 v16bf;
typedef __attribute__((ext_vector_type(8)))  float  v8f;

#define NEG_SLOPE 0.2f
#define EPS_F 1e-16f

// ---------- helpers ----------
__device__ __forceinline__ unsigned int enc_ord(float f) {
    unsigned int u = __float_as_uint(f);
    return (u & 0x80000000u) ? ~u : (u | 0x80000000u);
}
__device__ __forceinline__ float dec_ord(unsigned int u) {
    u = (u & 0x80000000u) ? (u & 0x7FFFFFFFu) : ~u;
    return __uint_as_float(u);
}
__device__ __forceinline__ float lrelu(float v) {
    return v > 0.f ? v : NEG_SLOPE * v;
}

// ---------- GEMM1: H1[N,128] = X[N,128] @ W1[128,128] (bf16 WMMA, f32 acc) ----------
__global__ __launch_bounds__(256) void gat_gemm1_wmma(
    const float* __restrict__ X, const float* __restrict__ W,
    float* __restrict__ H, int Nrows)
{
    __shared__ __bf16 ldsW[128 * 128]; // ldsW[col*128 + k] = W[k,col], 32 KB
    for (int i = threadIdx.x; i < 128 * 128; i += 256) {
        int col = i >> 7, k = i & 127;
        ldsW[i] = (__bf16)W[k * 128 + col];
    }
    __syncthreads();

    int wave = threadIdx.x >> 5;
    int lane = threadIdx.x & 31;
    int rowBase = (blockIdx.x * 8 + wave) * 16;
    if (rowBase >= Nrows) return;            // wave-uniform: EXEC stays all-1s

    int half = lane >> 4;                    // 0: lanes 0-15, 1: lanes 16-31
    int lrow = lane & 15;
    const float* xrow = X + (size_t)(rowBase + lrow) * 128;

    v8f zero = {};
    v8f acc[8];
#pragma unroll
    for (int j = 0; j < 8; ++j) acc[j] = zero;

#pragma unroll
    for (int kc = 0; kc < 4; ++kc) {
        // A fragment, contiguous K order: lanes 0-15 hold K=k0..k0+15 (k0=kc*32),
        // lanes 16-31 hold K offset by 16.
        int k0 = kc * 32 + half * 16;
        v16bf a;
#pragma unroll
        for (int t = 0; t < 8; ++t) {
            a[t]     = (__bf16)xrow[k0 + t];
            a[t + 8] = (__bf16)xrow[k0 + 8 + t];
        }
#pragma unroll
        for (int j = 0; j < 8; ++j) {
            int col = j * 16 + lrow;
            // B fragment: lane holds column `col`, K = kc*32 + half*16 .. +15, contiguous in ldsW
            const v16bf* bp = (const v16bf*)&ldsW[col * 128 + kc * 32 + half * 16];
            v16bf b = *bp;
            acc[j] = __builtin_amdgcn_wmma_f32_16x16x32_bf16(
                false, a, false, b, (short)0, acc[j], false, false);
        }
    }
    // C/D layout: VGPR r, lane: (M = r + 8*half, N = lrow)
    float* out = H + (size_t)rowBase * 128;
#pragma unroll
    for (int j = 0; j < 8; ++j) {
        int col = j * 16 + lrow;
#pragma unroll
        for (int r = 0; r < 8; ++r)
            out[(size_t)(r + 8 * half) * 128 + col] = acc[j][r];
    }
}

// ---------- GEMM2: H2[N,40] = Hin[N,128] @ W2[128,40], cols padded to 48 ----------
__global__ __launch_bounds__(256) void gat_gemm2_wmma(
    const float* __restrict__ Hin, const float* __restrict__ W2,
    float* __restrict__ H2, int Nrows)
{
    __shared__ __bf16 ldsW[48 * 128]; // 12 KB, cols 40..47 zero
    for (int i = threadIdx.x; i < 48 * 128; i += 256) {
        int col = i >> 7, k = i & 127;
        ldsW[i] = (col < 40) ? (__bf16)W2[k * 40 + col] : (__bf16)0.f;
    }
    __syncthreads();

    int wave = threadIdx.x >> 5;
    int lane = threadIdx.x & 31;
    int rowBase = (blockIdx.x * 8 + wave) * 16;
    if (rowBase >= Nrows) return;

    int half = lane >> 4;
    int lrow = lane & 15;
    const float* hrow = Hin + (size_t)(rowBase + lrow) * 128;

    v8f zero = {};
    v8f acc[3];
#pragma unroll
    for (int j = 0; j < 3; ++j) acc[j] = zero;

#pragma unroll
    for (int kc = 0; kc < 4; ++kc) {
        int k0 = kc * 32 + half * 16;
        v16bf a;
#pragma unroll
        for (int t = 0; t < 8; ++t) {
            a[t]     = (__bf16)hrow[k0 + t];
            a[t + 8] = (__bf16)hrow[k0 + 8 + t];
        }
#pragma unroll
        for (int j = 0; j < 3; ++j) {
            int col = j * 16 + lrow;
            const v16bf* bp = (const v16bf*)&ldsW[col * 128 + kc * 32 + half * 16];
            v16bf b = *bp;
            acc[j] = __builtin_amdgcn_wmma_f32_16x16x32_bf16(
                false, a, false, b, (short)0, acc[j], false, false);
        }
    }
#pragma unroll
    for (int j = 0; j < 3; ++j) {
        int col = j * 16 + lrow;
        if (col < 40) {
#pragma unroll
            for (int r = 0; r < 8; ++r)
                H2[(size_t)(rowBase + r + 8 * half) * 40 + col] = acc[j][r];
        }
    }
}

// ---------- per-node attention coefficients ----------
__global__ void gat_alpha1(const float* __restrict__ h1,
                           const float* __restrict__ a_src, const float* __restrict__ a_dst,
                           float* __restrict__ as, float* __restrict__ ad, int NH)
{
    int gid = blockIdx.x * blockDim.x + threadIdx.x;
    if (gid >= NH) return;
    int n = gid >> 3, h = gid & 7;
    const float* hp = h1 + (size_t)n * 128 + h * 16;
    const float* sv = a_src + h * 16;
    const float* dv = a_dst + h * 16;
    float accs = 0.f, accd = 0.f;
#pragma unroll
    for (int c = 0; c < 16; ++c) {
        float v = hp[c];
        accs += v * sv[c];
        accd += v * dv[c];
    }
    as[gid] = accs;
    ad[gid] = accd;
}

__global__ void gat_alpha2(const float* __restrict__ h2,
                           const float* __restrict__ a_src, const float* __restrict__ a_dst,
                           float* __restrict__ as, float* __restrict__ ad, int N)
{
    int n = blockIdx.x * blockDim.x + threadIdx.x;
    if (n >= N) return;
    const float* hp = h2 + (size_t)n * 40;
    float accs = 0.f, accd = 0.f;
#pragma unroll
    for (int c = 0; c < 40; ++c) {
        float v = hp[c];
        accs += v * a_src[c];
        accd += v * a_dst[c];
    }
    as[n] = accs;
    ad[n] = accd;
}

// ---------- edge passes, layer 1 (8 heads) ----------
__global__ void gat_edge_max1(const int* __restrict__ src, const int* __restrict__ dst,
                              const float* __restrict__ as, const float* __restrict__ ad,
                              unsigned int* __restrict__ m, int EH)
{
    int gid = blockIdx.x * blockDim.x + threadIdx.x;
    if (gid >= EH) return;
    int e = gid >> 3, h = gid & 7;
    int s = src[e], d = dst[e];
    float v = lrelu(as[s * 8 + h] + ad[d * 8 + h]);
    atomicMax(&m[d * 8 + h], enc_ord(v));
}

__global__ void gat_edge_sum1(const int* __restrict__ src, const int* __restrict__ dst,
                              const float* __restrict__ as, const float* __restrict__ ad,
                              const unsigned int* __restrict__ m, float* __restrict__ den, int EH)
{
    int gid = blockIdx.x * blockDim.x + threadIdx.x;
    if (gid >= EH) return;
    int e = gid >> 3, h = gid & 7;
    int s = src[e], d = dst[e];
    float v = lrelu(as[s * 8 + h] + ad[d * 8 + h]);
    atomicAdd(&den[d * 8 + h], __expf(v - dec_ord(m[d * 8 + h])));
}

__global__ void gat_edge_agg1(const int* __restrict__ src, const int* __restrict__ dst,
                              const float* __restrict__ as, const float* __restrict__ ad,
                              const unsigned int* __restrict__ m, const float* __restrict__ den,
                              const float* __restrict__ h1, float* __restrict__ acc, int EC)
{
    int gid = blockIdx.x * blockDim.x + threadIdx.x;
    if (gid >= EC) return;
    int e = gid >> 7, c = gid & 127, h = c >> 4;
    int s = src[e], d = dst[e];
    float v = lrelu(as[s * 8 + h] + ad[d * 8 + h]);
    float alpha = __expf(v - dec_ord(m[d * 8 + h])) / (den[d * 8 + h] + EPS_F);
    atomicAdd(&acc[(size_t)d * 128 + c], alpha * h1[(size_t)s * 128 + c]);
}

// ---------- ELU(acc + b1) in place ----------
__global__ void gat_elu_bias(float* __restrict__ acc, const float* __restrict__ b1, int NC)
{
    int gid = blockIdx.x * blockDim.x + threadIdx.x;
    if (gid >= NC) return;
    float v = acc[gid] + b1[gid & 127];
    acc[gid] = v > 0.f ? v : (__expf(v) - 1.f);
}

// ---------- edge passes, layer 2 (1 head, 40 channels) ----------
__global__ void gat_edge_max2(const int* __restrict__ src, const int* __restrict__ dst,
                              const float* __restrict__ as, const float* __restrict__ ad,
                              unsigned int* __restrict__ m, int E)
{
    int e = blockIdx.x * blockDim.x + threadIdx.x;
    if (e >= E) return;
    float v = lrelu(as[src[e]] + ad[dst[e]]);
    atomicMax(&m[dst[e]], enc_ord(v));
}

__global__ void gat_edge_sum2(const int* __restrict__ src, const int* __restrict__ dst,
                              const float* __restrict__ as, const float* __restrict__ ad,
                              const unsigned int* __restrict__ m, float* __restrict__ den, int E)
{
    int e = blockIdx.x * blockDim.x + threadIdx.x;
    if (e >= E) return;
    int d = dst[e];
    float v = lrelu(as[src[e]] + ad[d]);
    atomicAdd(&den[d], __expf(v - dec_ord(m[d])));
}

__global__ void gat_edge_agg2(const int* __restrict__ src, const int* __restrict__ dst,
                              const float* __restrict__ as, const float* __restrict__ ad,
                              const unsigned int* __restrict__ m, const float* __restrict__ den,
                              const float* __restrict__ h2, float* __restrict__ out, int E64)
{
    int gid = blockIdx.x * blockDim.x + threadIdx.x;
    if (gid >= E64) return;
    int e = gid >> 6, c = gid & 63;
    if (c >= 40) return;
    int s = src[e], d = dst[e];
    float v = lrelu(as[s] + ad[d]);
    float alpha = __expf(v - dec_ord(m[d])) / (den[d] + EPS_F);
    atomicAdd(&out[(size_t)d * 40 + c], alpha * h2[(size_t)s * 40 + c]);
}

// ---------- bias + log_softmax over 40 classes, in place on out ----------
__global__ void gat_logsoftmax(float* __restrict__ out, const float* __restrict__ b2, int N)
{
    int n = blockIdx.x * blockDim.x + threadIdx.x;
    if (n >= N) return;
    float v[40];
    float mx = -INFINITY;
#pragma unroll
    for (int c = 0; c < 40; ++c) {
        v[c] = out[(size_t)n * 40 + c] + b2[c];
        mx = fmaxf(mx, v[c]);
    }
    float s = 0.f;
#pragma unroll
    for (int c = 0; c < 40; ++c) s += __expf(v[c] - mx);
    float l = mx + logf(s);
#pragma unroll
    for (int c = 0; c < 40; ++c) out[(size_t)n * 40 + c] = v[c] - l;
}

extern "C" void kernel_launch(void* const* d_in, const int* in_sizes, int n_in,
                              void* d_out, int out_size, void* d_ws, size_t ws_size,
                              hipStream_t stream)
{
    const float* x     = (const float*)d_in[0];
    const int*   eidx  = (const int*)  d_in[1];
    const float* W1    = (const float*)d_in[2];
    const float* asr1  = (const float*)d_in[3];
    const float* ads1  = (const float*)d_in[4];
    const float* b1    = (const float*)d_in[5];
    const float* W2    = (const float*)d_in[6];
    const float* asr2  = (const float*)d_in[7];
    const float* ads2  = (const float*)d_in[8];
    const float* b2    = (const float*)d_in[9];
    float* out = (float*)d_out;

    const int N = in_sizes[0] / 128;   // 100000
    const int E = in_sizes[1] / 2;     // 1600000
    const int* src = eidx;
    const int* dst = eidx + E;

    // workspace layout (floats)
    float* ws = (float*)d_ws;
    size_t o = 0;
    float* h1   = ws + o; o += (size_t)N * 128;
    float* acc1 = ws + o; o += (size_t)N * 128;   // becomes h after ELU
    float* h2   = ws + o; o += (size_t)N * 40;
    float* as1  = ws + o; o += (size_t)N * 8;
    float* ad1  = ws + o; o += (size_t)N * 8;
    unsigned int* m1 = (unsigned int*)(ws + o); o += (size_t)N * 8;
    float* dn1  = ws + o; o += (size_t)N * 8;
    float* as2  = ws + o; o += (size_t)N;
    float* ad2  = ws + o; o += (size_t)N;
    unsigned int* m2 = (unsigned int*)(ws + o); o += (size_t)N;
    float* dn2  = ws + o; o += (size_t)N;
    (void)ws_size; (void)n_in; (void)out_size;

    // zero accumulators (enc_ord identity for max is 0)
    hipMemsetAsync(acc1, 0, (size_t)N * 128 * sizeof(float), stream);
    hipMemsetAsync(m1,   0, (size_t)N * 8 * 2 * sizeof(float), stream); // m1 + dn1
    hipMemsetAsync(m2,   0, (size_t)N * 2 * sizeof(float), stream);     // m2 + dn2
    hipMemsetAsync(out,  0, (size_t)N * 40 * sizeof(float), stream);

    const int T = 256;
    int gemmBlocks = ((N / 16) + 7) / 8;

    // layer 1
    gat_gemm1_wmma<<<gemmBlocks, T, 0, stream>>>(x, W1, h1, N);
    gat_alpha1<<<(N * 8 + T - 1) / T, T, 0, stream>>>(h1, asr1, ads1, as1, ad1, N * 8);
    gat_edge_max1<<<(E * 8 + T - 1) / T, T, 0, stream>>>(src, dst, as1, ad1, m1, E * 8);
    gat_edge_sum1<<<(E * 8 + T - 1) / T, T, 0, stream>>>(src, dst, as1, ad1, m1, dn1, E * 8);
    {
        long long EC = (long long)E * 128;
        gat_edge_agg1<<<(unsigned)((EC + T - 1) / T), T, 0, stream>>>(
            src, dst, as1, ad1, m1, dn1, h1, acc1, (int)EC);
    }
    gat_elu_bias<<<(N * 128 + T - 1) / T, T, 0, stream>>>(acc1, b1, N * 128);

    // layer 2
    gat_gemm2_wmma<<<gemmBlocks, T, 0, stream>>>(acc1, W2, h2, N);
    gat_alpha2<<<(N + T - 1) / T, T, 0, stream>>>(h2, asr2, ads2, as2, ad2, N);
    gat_edge_max2<<<(E + T - 1) / T, T, 0, stream>>>(src, dst, as2, ad2, m2, E);
    gat_edge_sum2<<<(E + T - 1) / T, T, 0, stream>>>(src, dst, as2, ad2, m2, dn2, E);
    {
        long long E64 = (long long)E * 64;
        gat_edge_agg2<<<(unsigned)((E64 + T - 1) / T), T, 0, stream>>>(
            src, dst, as2, ad2, m2, dn2, h2, out, (int)E64);
    }
    gat_logsoftmax<<<(N + T - 1) / T, T, 0, stream>>>(out, b2, N);
}
